// SinkBF_binary_70403103916196
// MI455X (gfx1250) — matrix operand order repaired and checked by reading, not compiled, not run on hardware
//
#include <hip/hip_runtime.h>
#include <hip/hip_bf16.h>

// ---------------------------------------------------------------------------
// Problem dims (fixed by reference)
// ---------------------------------------------------------------------------
#define NXD   512
#define NWD   512
#define NYD   256
#define NB    1024
#define NWUPD 128
#define NHR   32

typedef __attribute__((ext_vector_type(16))) __bf16       v16bf;
typedef __attribute__((ext_vector_type(8)))  float        v8f;
typedef __attribute__((ext_vector_type(4)))  unsigned int v4u;

// LDS tile geometry: 64 rows x 64 bf16, padded to 72 bf16 (144B) per row so
// 16-lane fragment reads stride 36 dwords -> all 64 banks, conflict-free.
#define TPAD 72

// RNE float -> bf16
__device__ __forceinline__ __bf16 f2bf(float f) {
    unsigned u = __builtin_bit_cast(unsigned, f);
    u += 0x7FFFu + ((u >> 16) & 1u);
    unsigned short s = (unsigned short)(u >> 16);
    return __builtin_bit_cast(__bf16, s);
}

// ---------------------------------------------------------------------------
// WMMA fragment loaders from LDS (wave32, v_wmma_f32_16x16x32_bf16)
// A (16x32 MxK): lane l: row M=(l&15); K chunks [kh..kh+7] and [16+kh..16+kh+7],
//                kh = (l>=16)*8  -> two 16B ds loads
// B (32x16 KxN) from row-major W[N,K]: lane l: col N=(l&15); K chunks
//                [kh..kh+7] and [16+kh..16+kh+7], kh = (l>=16)*8 -> two 16B ds loads
// ---------------------------------------------------------------------------
__device__ __forceinline__ v16bf lds_fragA(const __bf16* tile, int row, int ksub, int lane) {
    const __bf16* p = tile + (size_t)(row + (lane & 15)) * TPAD + ksub + ((lane >> 4) << 3);
    union { v16bf f; v4u u[2]; } t;
    t.u[0] = *(const v4u*)p;
    t.u[1] = *(const v4u*)(p + 16);
    return t.f;
}

__device__ __forceinline__ v16bf lds_fragB(const __bf16* tile, int col, int ksub, int lane) {
    const __bf16* p = tile + (size_t)(col + (lane & 15)) * TPAD + ksub + ((lane >> 4) << 3);
    union { v16bf f; v4u u[2]; } t;
    t.u[0] = *(const v4u*)p;
    t.u[1] = *(const v4u*)(p + 16);
    return t.f;
}

// ---------------------------------------------------------------------------
// Async stage loader: move one 64x64 bf16 tile of A and of W into LDS with
// GLOBAL_LOAD_ASYNC_TO_LDS_B128 (each lane moves one 16B chunk; 4 instr per
// wave per matrix per stage -> ASYNCcnt 8 per wave per stage).
// ---------------------------------------------------------------------------
__device__ __forceinline__ void stage_load(const __bf16* __restrict__ A,
                                           const __bf16* __restrict__ W,
                                           int K, int rowB, int colB, int k0,
                                           unsigned ldsA, unsigned ldsW, int tid)
{
#pragma unroll
    for (int i = 0; i < 4; ++i) {
        const int c  = i * 128 + tid;        // chunk 0..511 (16B each)
        const int r  = c >> 3;               // tile row 0..63
        const int ko = (c & 7) * 8;          // K offset within 64-slab
        const unsigned la = ldsA + (unsigned)(r * (TPAD * 2) + ko * 2);
        const unsigned lw = ldsW + (unsigned)(r * (TPAD * 2) + ko * 2);
        const unsigned long long ga =
            (unsigned long long)(size_t)(A + (size_t)(rowB + r) * K + k0 + ko);
        const unsigned long long gw =
            (unsigned long long)(size_t)(W + (size_t)(colB + r) * K + k0 + ko);
        asm volatile("global_load_async_to_lds_b128 %0, %1, off"
                     :: "v"(la), "v"(ga) : "memory");
        asm volatile("global_load_async_to_lds_b128 %0, %1, off"
                     :: "v"(lw), "v"(gw) : "memory");
    }
}

// ---------------------------------------------------------------------------
// GEMM:  out = epi( A[MxK] * W[NxK]^T + bias[N] + resid[MxN] )
// epi: 0=identity, 1=tanh, 2=logit (outF raw, outB = tanh(0.5*x))
// 128 threads (4 waves), 64x64 block tile, double-buffered K=64 LDS stages.
// ---------------------------------------------------------------------------
__global__ void __launch_bounds__(128)
k_gemm(const __bf16* __restrict__ A, const __bf16* __restrict__ W,
       const float* __restrict__ bias, const float* __restrict__ resid,
       float* __restrict__ outF, __bf16* __restrict__ outB,
       int M, int N, int K, int epi)
{
    __shared__ __align__(16) __bf16 shA[2][64 * TPAD];
    __shared__ __align__(16) __bf16 shW[2][64 * TPAD];
    (void)M;

    const int tid  = threadIdx.x;
    const int lane = tid & 31;
    const int wid  = tid >> 5;
    const int rowB = blockIdx.y * 64;
    const int colB = blockIdx.x * 64;
    const int wm   = (wid & 1) * 32;       // wave row offset inside tile
    const int wn   = (wid >> 1) * 32;      // wave col offset inside tile

    const unsigned bA0 = (unsigned)(size_t)&shA[0][0];
    const unsigned bA1 = (unsigned)(size_t)&shA[1][0];
    const unsigned bW0 = (unsigned)(size_t)&shW[0][0];
    const unsigned bW1 = (unsigned)(size_t)&shW[1][0];

    const v8f vzero = {0.f, 0.f, 0.f, 0.f, 0.f, 0.f, 0.f, 0.f};
    v8f acc[2][2];
    acc[0][0] = vzero; acc[0][1] = vzero; acc[1][0] = vzero; acc[1][1] = vzero;

    stage_load(A, W, K, rowB, colB, 0, bA0, bW0, tid);

    const int nk = K >> 6;                 // K is a multiple of 64 here
    for (int s = 0; s < nk; ++s) {
        const int buf = s & 1;
        if (s + 1 < nk) {
            stage_load(A, W, K, rowB, colB, (s + 1) << 6,
                       buf ? bA0 : bA1, buf ? bW0 : bW1, tid);
            asm volatile("s_wait_asynccnt 8" ::: "memory");  // stage s done (FIFO)
        } else {
            asm volatile("s_wait_asynccnt 0" ::: "memory");  // drain
        }
        __syncthreads();                                     // publish stage s

        const __bf16* tA = shA[buf];
        const __bf16* tW = shW[buf];
#pragma unroll
        for (int ks = 0; ks < 64; ks += 32) {
            v16bf a0 = lds_fragA(tA, wm +  0, ks, lane);
            v16bf a1 = lds_fragA(tA, wm + 16, ks, lane);
            v16bf b0 = lds_fragB(tW, wn +  0, ks, lane);
            v16bf b1 = lds_fragB(tW, wn + 16, ks, lane);
            acc[0][0] = __builtin_amdgcn_wmma_f32_16x16x32_bf16(false, a0, false, b0, (short)0, acc[0][0], false, false);
            acc[0][1] = __builtin_amdgcn_wmma_f32_16x16x32_bf16(false, a0, false, b1, (short)0, acc[0][1], false, false);
            acc[1][0] = __builtin_amdgcn_wmma_f32_16x16x32_bf16(false, a1, false, b0, (short)0, acc[1][0], false, false);
            acc[1][1] = __builtin_amdgcn_wmma_f32_16x16x32_bf16(false, a1, false, b1, (short)0, acc[1][1], false, false);
        }
        __syncthreads();                   // all waves done reading buf before overwrite
    }

    // C/D layout: VGPR j -> row (lane>=16 ? 8+j : j), col = lane&15
    const int rowW  = rowB + wm;
    const int colW  = colB + wn;
    const int nOff  = lane & 15;
    const int mBase = (lane >> 4) << 3;
#pragma unroll
    for (int ti = 0; ti < 2; ++ti) {
#pragma unroll
        for (int tj = 0; tj < 2; ++tj) {
#pragma unroll
            for (int j = 0; j < 8; ++j) {
                const int gm = rowW + ti * 16 + mBase + j;
                const int gn = colW + tj * 16 + nOff;
                float s = acc[ti][tj][j];
                if (bias)  s += bias[gn];
                if (resid) s += resid[(size_t)gm * N + gn];
                float vF = s, vB = s;
                if (epi == 1)      { vF = tanhf(s); vB = vF; }
                else if (epi == 2) { vB = tanhf(0.5f * s); }
                const size_t o = (size_t)gm * N + gn;
                if (outF) outF[o] = vF;
                if (outB) outB[o] = f2bf(vB);
            }
        }
    }
}

// ---------------------------------------------------------------------------
// Fused GRU cell elementwise (biases already folded into gi/gh by the GEMMs)
// ---------------------------------------------------------------------------
__global__ void k_gru(const float* __restrict__ gi, const float* __restrict__ gh,
                      const float* __restrict__ h, float* __restrict__ h2,
                      __bf16* __restrict__ h2bf)
{
    const size_t i = (size_t)blockIdx.x * blockDim.x + threadIdx.x; // < NB*NXD
    const int b = (int)(i >> 9);
    const int j = (int)(i & 511);
    const size_t g = (size_t)b * (3 * NXD) + j;
    const float ir = gi[g], iz = gi[g + NXD], in = gi[g + 2 * NXD];
    const float hr = gh[g], hz = gh[g + NXD], hn = gh[g + 2 * NXD];
    const float r  = 1.f / (1.f + __expf(-(ir + hr)));
    const float z  = 1.f / (1.f + __expf(-(iz + hz)));
    const float n  = tanhf(in + r * hn);
    const float o  = (1.f - z) * n + z * h[i];
    h2[i]   = o;
    h2bf[i] = f2bf(o);
}

// Deterministic +/-1 bf16 inputs for the free-run phase
__global__ void k_rand(__bf16* __restrict__ w, unsigned seed)
{
    const unsigned i = blockIdx.x * blockDim.x + threadIdx.x;
    unsigned x = i * 2654435761u + seed;
    x ^= x >> 16; x *= 0x7FEB352Du;
    x ^= x >> 15; x *= 0x846CA68Bu;
    x ^= x >> 16;
    w[i] = (x & 1u) ? f2bf(-1.0f) : f2bf(1.0f);
}

__global__ void k_cvt(const float* __restrict__ s, __bf16* __restrict__ d, size_t n)
{
    const size_t i = (size_t)blockIdx.x * blockDim.x + threadIdx.x;
    if (i < n) d[i] = f2bf(s[i]);
}

// Strided column extraction + convert (splitting Wl1 into h / y parts)
__global__ void k_extract(const float* __restrict__ s, __bf16* __restrict__ d,
                          int rows, int ld, int off, int cols)
{
    const size_t i = (size_t)blockIdx.x * blockDim.x + threadIdx.x;
    if (i < (size_t)rows * cols) {
        const int r = (int)(i / cols);
        const int c = (int)(i % cols);
        d[i] = f2bf(s[(size_t)r * ld + off + c]);
    }
}

// ---------------------------------------------------------------------------
// Host orchestration
// ---------------------------------------------------------------------------
extern "C" void kernel_launch(void* const* d_in, const int* in_sizes, int n_in,
                              void* d_out, int out_size, void* d_ws, size_t ws_size,
                              hipStream_t stream)
{
    (void)in_sizes; (void)n_in; (void)out_size; (void)ws_size;

    const float* Y0  = (const float*)d_in[0];
    // d_in[1] = Nhrzn scalar (compile-time NHR)
    const float* Wq  = (const float*)d_in[2];
    const float* bq  = (const float*)d_in[3];
    const float* Wp1 = (const float*)d_in[4];
    const float* bp1 = (const float*)d_in[5];
    const float* Wp2 = (const float*)d_in[6];
    const float* bp2 = (const float*)d_in[7];
    const float* Wl1 = (const float*)d_in[8];
    const float* bl1 = (const float*)d_in[9];
    const float* Wl2 = (const float*)d_in[10];
    const float* bl2 = (const float*)d_in[11];
    const float* Wih = (const float*)d_in[12];
    const float* Whh = (const float*)d_in[13];
    const float* bih = (const float*)d_in[14];
    const float* bhh = (const float*)d_in[15];

    const int T = NWUPD + 1 + NHR;                       // 161
    float* Xout = (float*)d_out;                         // [161,1024,512]
    float* Lout = Xout + (size_t)T * NB * NXD;           // [128,1024,512]
    float* Yout = Lout + (size_t)NWUPD * NB * NWD;       // [161,1024,256]

    // Workspace bump allocator (256B aligned)
    char* wsp = (char*)d_ws;
    auto alloc = [&](size_t bytes) -> void* {
        void* p = (void*)wsp;
        wsp += (bytes + 255) & ~(size_t)255;
        return p;
    };
    __bf16* Wq_b   = (__bf16*)alloc((size_t)NXD * NYD * 2);
    __bf16* Wl1h_b = (__bf16*)alloc((size_t)NWD * NXD * 2);
    __bf16* Wl1y_b = (__bf16*)alloc((size_t)NWD * NYD * 2);
    __bf16* Wl2_b  = (__bf16*)alloc((size_t)NWD * NWD * 2);
    __bf16* Wih_b  = (__bf16*)alloc((size_t)3 * NXD * NWD * 2);
    __bf16* Whh_b  = (__bf16*)alloc((size_t)3 * NXD * NXD * 2);
    __bf16* Wp1_b  = (__bf16*)alloc((size_t)NXD * NXD * 2);
    __bf16* Wp2_b  = (__bf16*)alloc((size_t)NYD * NXD * 2);
    __bf16* y_b    = (__bf16*)alloc((size_t)NB * NYD * 2);
    __bf16* h_b    = (__bf16*)alloc((size_t)NB * NXD * 2);
    __bf16* T1_b   = (__bf16*)alloc((size_t)NB * NWD * 2);
    __bf16* w_b    = (__bf16*)alloc((size_t)NB * NWD * 2);
    float*  tmpL   = (float*)alloc((size_t)NB * NWD * 4);
    float*  gi     = (float*)alloc((size_t)NB * 3 * NXD * 4);
    float*  gh     = (float*)alloc((size_t)NB * 3 * NXD * 4);
    const int CH = 16;                                   // final-proj chunk (timesteps)
    __bf16* Xc_b   = (__bf16*)alloc((size_t)CH * NB * NXD * 2);
    __bf16* Tc_b   = (__bf16*)alloc((size_t)CH * NB * NXD * 2);

    auto gemm = [&](const __bf16* Am, const __bf16* Wm, const float* bias,
                    const float* resid, float* oF, __bf16* oB,
                    int M, int N, int K, int epi) {
        k_gemm<<<dim3(N / 64, M / 64), dim3(128), 0, stream>>>(
            Am, Wm, bias, resid, oF, oB, M, N, K, epi);
    };
    auto cvt = [&](const float* s, __bf16* d, size_t n) {
        k_cvt<<<dim3((unsigned)((n + 255) / 256)), dim3(256), 0, stream>>>(s, d, n);
    };

    // --- weight precompute (bf16 copies; Wl1 split into h / y column blocks)
    cvt(Wq,  Wq_b,  (size_t)NXD * NYD);
    k_extract<<<dim3((NWD * NXD + 255) / 256), dim3(256), 0, stream>>>(Wl1, Wl1h_b, NWD, NXD + NYD, 0,   NXD);
    k_extract<<<dim3((NWD * NYD + 255) / 256), dim3(256), 0, stream>>>(Wl1, Wl1y_b, NWD, NXD + NYD, NXD, NYD);
    cvt(Wl2, Wl2_b, (size_t)NWD * NWD);
    cvt(Wih, Wih_b, (size_t)3 * NXD * NWD);
    cvt(Whh, Whh_b, (size_t)3 * NXD * NXD);
    cvt(Wp1, Wp1_b, (size_t)NXD * NXD);
    cvt(Wp2, Wp2_b, (size_t)NYD * NXD);

    const size_t SLAB = (size_t)NB * NXD;   // one timestep of X
    const size_t YSL  = (size_t)NB * NYD;   // one timestep of Y0 / Yhat

    // --- x0 = tanh(Y0[0] @ Wq^T + bq) -> X[0] (fp32) and h_b (bf16)
    cvt(Y0, y_b, YSL);
    gemm(y_b, Wq_b, bq, nullptr, Xout, h_b, NB, NXD, NYD, /*tanh*/1);

    // --- warm-up scan (128 steps)
    for (int t = 0; t < NWUPD; ++t) {
        cvt(Y0 + (size_t)(1 + t) * YSL, y_b, YSL);
        // tmpL = y @ Wl1y^T
        gemm(y_b, Wl1y_b, nullptr, nullptr, tmpL, nullptr, NB, NWD, NYD, 0);
        // T1 = tanh(h @ Wl1h^T + tmpL + bl1)
        gemm(h_b, Wl1h_b, bl1, tmpL, nullptr, T1_b, NB, NWD, NXD, 1);
        // logit = T1 @ Wl2^T + bl2 ; w = tanh(0.5*logit)
        gemm(T1_b, Wl2_b, bl2, nullptr, Lout + (size_t)t * NB * NWD, w_b, NB, NWD, NWD, 2);
        // gi = w @ W_ih^T + b_ih ; gh = h @ W_hh^T + b_hh
        gemm(w_b, Wih_b, bih, nullptr, gi, nullptr, NB, 3 * NXD, NWD, 0);
        gemm(h_b, Whh_b, bhh, nullptr, gh, nullptr, NB, 3 * NXD, NXD, 0);
        // h2 = GRU(gi, gh, h) -> X[t+1] and h_b
        k_gru<<<dim3((unsigned)(SLAB / 256)), dim3(256), 0, stream>>>(
            gi, gh, Xout + (size_t)t * SLAB, Xout + (size_t)(t + 1) * SLAB, h_b);
    }

    // --- free-run scan (32 steps, deterministic +/-1 inputs)
    for (int t = 0; t < NHR; ++t) {
        k_rand<<<dim3((unsigned)((size_t)NB * NWD / 256)), dim3(256), 0, stream>>>(
            w_b, 0x85EBCA6Bu + 0x9E3779B9u * (unsigned)t);
        gemm(w_b, Wih_b, bih, nullptr, gi, nullptr, NB, 3 * NXD, NWD, 0);
        gemm(h_b, Whh_b, bhh, nullptr, gh, nullptr, NB, 3 * NXD, NXD, 0);
        k_gru<<<dim3((unsigned)(SLAB / 256)), dim3(256), 0, stream>>>(
            gi, gh, Xout + (size_t)(NWUPD + t) * SLAB,
            Xout + (size_t)(NWUPD + 1 + t) * SLAB, h_b);
    }

    // --- Yhat = tanh(X @ Wp1^T + bp1) @ Wp2^T + bp2, chunked over time
    for (int t0 = 0; t0 < T; t0 += CH) {
        const int ts = (T - t0 < CH) ? (T - t0) : CH;
        const int Mrows = ts * NB;
        cvt(Xout + (size_t)t0 * SLAB, Xc_b, (size_t)Mrows * NXD);
        gemm(Xc_b, Wp1_b, bp1, nullptr, nullptr, Tc_b, Mrows, NXD, NXD, 1);
        gemm(Tc_b, Wp2_b, bp2, nullptr, Yout + (size_t)t0 * YSL, nullptr, Mrows, NYD, NXD, 0);
    }
}
